// MultiHeadAttention_43276090475158
// MI455X (gfx1250) — compile-verified, matrix operand order checked by reading, not run on hardware
//
#include <hip/hip_runtime.h>
#include <hip/hip_bf16.h>

// ---------------------------------------------------------------------------
// MultiHeadAttention for MI455X (gfx1250): bf16 WMMA everywhere, f32 accum.
// B=4, S=2048, D=768, H=12, DK=64.
// Round 3: defer the softmax l-reduction out of the kv loop (per-lane partial
// sums; single DPP rowsum at the end). Inner loop now has only the row-max
// DPP reduction + exps competing with the WMMAs.
// ---------------------------------------------------------------------------

#define B_   4
#define S_   2048
#define D_   768
#define H_   12
#define DK_  64
#define R_   (B_ * S_)     // 8192 total rows
#define BH_  (B_ * H_)     // 48 head-batches

typedef __bf16 bf16_t;
typedef __attribute__((ext_vector_type(16))) __bf16 v16bf;
typedef __attribute__((ext_vector_type(8)))  __bf16 v8bf;
typedef __attribute__((ext_vector_type(8)))  float  v8f;
typedef __attribute__((ext_vector_type(4)))  float  v4f;

// ---- WMMA fragment helpers -------------------------------------------------
// A-fragment (16x32 bf16): lane&15 = row M; lane>>4 selects K-subpattern:
//   half 0: K = {0..7, 16..23}, half 1: K = {8..15, 24..31}.
// Elements 0..7  <- p[0..7], elements 8..15 <- p[16..23]
// where p = rowBase + (lane>>4)*8.  B-fragment uses the symmetric layout with
// lane&15 = column N, same K packing, so "load the N-th row of B^T" works.

static __device__ inline v16bf frag_bf16(const bf16_t* p) {
    v8bf lo = *(const v8bf*)(p);
    v8bf hi = *(const v8bf*)(p + 16);
    v16bf r;
#pragma unroll
    for (int i = 0; i < 8; ++i) { r[i] = lo[i]; r[i + 8] = hi[i]; }
    return r;
}

static __device__ inline v16bf frag_f32(const float* p) {
    v4f a0 = *(const v4f*)(p);
    v4f a1 = *(const v4f*)(p + 4);
    v4f b0 = *(const v4f*)(p + 16);
    v4f b1 = *(const v4f*)(p + 20);
    v16bf r;
#pragma unroll
    for (int i = 0; i < 4; ++i) {
        r[i]      = (bf16_t)a0[i];
        r[i + 4]  = (bf16_t)a1[i];
        r[i + 8]  = (bf16_t)b0[i];
        r[i + 12] = (bf16_t)b1[i];
    }
    return r;
}

static __device__ inline v8f wmma_bf16(v16bf a, v16bf b, v8f c) {
    // D = A(16x32) * B(32x16) + C, f32 accumulate.
    return __builtin_amdgcn_wmma_f32_16x16x32_bf16(
        /*neg_a=*/false, a, /*neg_b=*/false, b,
        /*c_mod=*/(short)0, c, /*reuse_a=*/false, /*reuse_b=*/false);
}

// ---- 16-lane (DPP row) circular reductions: VALU-only, no LDS traffic -----
// row_ror:n dpp_ctrl = 0x120|n. After combining with ror 1,2,4,8 every lane
// of the 16-lane row holds the full-row result. Wave32 halves reduce
// independently, which matches the C/D layout (rows 0-7 vs 8-15).
static __device__ inline float rowmax16(float x) {
    int t;
    t = __builtin_amdgcn_update_dpp(0, __float_as_int(x), 0x121, 0xf, 0xf, true);
    x = fmaxf(x, __int_as_float(t));
    t = __builtin_amdgcn_update_dpp(0, __float_as_int(x), 0x122, 0xf, 0xf, true);
    x = fmaxf(x, __int_as_float(t));
    t = __builtin_amdgcn_update_dpp(0, __float_as_int(x), 0x124, 0xf, 0xf, true);
    x = fmaxf(x, __int_as_float(t));
    t = __builtin_amdgcn_update_dpp(0, __float_as_int(x), 0x128, 0xf, 0xf, true);
    x = fmaxf(x, __int_as_float(t));
    return x;
}

static __device__ inline float rowsum16(float x) {
    int t;
    t = __builtin_amdgcn_update_dpp(0, __float_as_int(x), 0x121, 0xf, 0xf, true);
    x += __int_as_float(t);
    t = __builtin_amdgcn_update_dpp(0, __float_as_int(x), 0x122, 0xf, 0xf, true);
    x += __int_as_float(t);
    t = __builtin_amdgcn_update_dpp(0, __float_as_int(x), 0x124, 0xf, 0xf, true);
    x += __int_as_float(t);
    t = __builtin_amdgcn_update_dpp(0, __float_as_int(x), 0x128, 0xf, 0xf, true);
    x += __int_as_float(t);
    return x;
}

// ---- fp32 -> bf16 streaming convert (weights) ------------------------------
__global__ void cvt_f32_to_bf16(const float* __restrict__ in,
                                bf16_t* __restrict__ out, int n) {
    int i = blockIdx.x * blockDim.x + threadIdx.x;
    if (i < n) out[i] = (bf16_t)in[i];
}

// ---- 64x64-tile GEMM: Y = X @ W^T + bias -----------------------------------
// MODE 0: X fp32, write Q bf16 (BH,S,DK), scaled by 1/sqrt(DK)
// MODE 1: X fp32, write K bf16 (BH,S,DK)
// MODE 2: X fp32, write V^T bf16 (BH,DK,S)
// MODE 3: X bf16 (context), write fp32 row-major (final output)
template <int MODE>
__global__ __launch_bounds__(128) void gemm64(const void* __restrict__ Ap,
                                              const bf16_t* __restrict__ Wb,
                                              const float* __restrict__ bias,
                                              void* __restrict__ Yp) {
    const int lane = threadIdx.x & 31;
    const int wave = threadIdx.x >> 5;
    const int l16  = lane & 15;
    const int hs   = (lane >> 4) & 1;
    const int mrow = blockIdx.x * 64 + wave * 16 + l16;  // A row for this lane
    const int n0   = blockIdx.y * 64;

    const float*  Af = (const float*)Ap;
    const bf16_t* Ab = (const bf16_t*)Ap;

    v8f acc[4] = {};
#pragma unroll 1
    for (int k0 = 0; k0 < D_; k0 += 32) {
        v16bf a = (MODE == 3)
                      ? frag_bf16(Ab + (size_t)mrow * D_ + k0 + hs * 8)
                      : frag_f32 (Af + (size_t)mrow * D_ + k0 + hs * 8);
#pragma unroll
        for (int nb = 0; nb < 4; ++nb) {
            const bf16_t* wp =
                Wb + (size_t)(n0 + nb * 16 + l16) * D_ + k0 + hs * 8;
            if (k0 + 32 < D_) __builtin_prefetch(wp + 32, 0, 1);
            v16bf bf = frag_bf16(wp);
            acc[nb] = wmma_bf16(a, bf, acc[nb]);
        }
    }

    // Epilogue. C/D layout: VGPR r -> row (r + hs*8), lane&15 -> column.
#pragma unroll
    for (int nb = 0; nb < 4; ++nb) {
        const int col = n0 + nb * 16 + l16;
        const float bv = bias[col];
#pragma unroll
        for (int r = 0; r < 8; ++r) {
            const int gr = blockIdx.x * 64 + wave * 16 + r + hs * 8;
            const float y = acc[nb][r] + bv;
            if (MODE == 0) {
                const int b = gr >> 11, s = gr & (S_ - 1);
                const int h = col >> 6, dk = col & 63;
                ((bf16_t*)Yp)[(((size_t)(b * H_ + h)) * S_ + s) * DK_ + dk] =
                    (bf16_t)(y * 0.125f);  // fold 1/sqrt(64)
            } else if (MODE == 1) {
                const int b = gr >> 11, s = gr & (S_ - 1);
                const int h = col >> 6, dk = col & 63;
                ((bf16_t*)Yp)[(((size_t)(b * H_ + h)) * S_ + s) * DK_ + dk] =
                    (bf16_t)y;
            } else if (MODE == 2) {
                const int b = gr >> 11, s = gr & (S_ - 1);
                const int h = col >> 6, dk = col & 63;
                ((bf16_t*)Yp)[(((size_t)(b * H_ + h)) * DK_ + dk) * S_ + s] =
                    (bf16_t)y;
            } else {
                ((float*)Yp)[(size_t)gr * D_ + col] = y;
            }
        }
    }
}

// ---- Flash attention: per-wave 16 query rows, kv tiles of 32 ---------------
__global__ __launch_bounds__(128) void attn64(const bf16_t* __restrict__ Qb,
                                              const bf16_t* __restrict__ Kb,
                                              const bf16_t* __restrict__ VTb,
                                              const int* __restrict__ mask,
                                              bf16_t* __restrict__ Ctx) {
    __shared__ __align__(16) bf16_t pbuf[4][16 * 32];  // per-wave P transpose

    const int lane = threadIdx.x & 31;
    const int wave = threadIdx.x >> 5;
    const int l16  = lane & 15;
    const int hs   = (lane >> 4) & 1;
    const int bh   = blockIdx.y;
    const int b    = bh / H_;
    const int h    = bh % H_;
    const int q0   = blockIdx.x * 64 + wave * 16;

    // Q A-fragments for all DK=64 (two K=32 chunks), kept resident.
    const bf16_t* qp = Qb + ((size_t)bh * S_ + q0 + l16) * DK_ + hs * 8;
    const v16bf qa0 = frag_bf16(qp);        // dk 0..31
    const v16bf qa1 = frag_bf16(qp + 32);   // dk 32..63

    // mi: running row max (row-uniform across the 16-lane group).
    // li: PER-LANE partial of l (columns this lane owns); the recurrence
    //     l = l*sc + rowsum(P) is linear and sc is row-uniform, so the
    //     cross-lane reduction commutes to a single rowsum16 at the end.
    float mi[8], li[8];
#pragma unroll
    for (int r = 0; r < 8; ++r) { mi[r] = -1e30f; li[r] = 0.0f; }
    v8f o[4] = {};  // O accumulator: 16 rows x 64 dk (4 N-blocks)

    bf16_t* pw = &pbuf[wave][0];

    for (int kv0 = 0; kv0 < S_; kv0 += 32) {
        // ---- scores S = Q K^T (two 16x16 N-blocks over kv tile of 32) ----
        const bf16_t* kp0 = Kb + ((size_t)bh * S_ + kv0 + l16) * DK_ + hs * 8;
        const bf16_t* kp1 = kp0 + 16 * DK_;
        v8f s0 = {}, s1 = {};
        s0 = wmma_bf16(qa0, frag_bf16(kp0),      s0);
        s0 = wmma_bf16(qa1, frag_bf16(kp0 + 32), s0);
        s1 = wmma_bf16(qa0, frag_bf16(kp1),      s1);
        s1 = wmma_bf16(qa1, frag_bf16(kp1 + 32), s1);

        // key-padding mask: column bias (column index == lane&15 per block)
        const float mb0 = (mask[b * S_ + kv0 + l16]      == 0) ? -1e9f : 0.0f;
        const float mb1 = (mask[b * S_ + kv0 + 16 + l16] == 0) ? -1e9f : 0.0f;

        // ---- online softmax; per-lane stats align with C/D row layout ----
#pragma unroll
        for (int r = 0; r < 8; ++r) {
            float a0 = s0[r] + mb0;
            float a1 = s1[r] + mb1;
            const float t  = rowmax16(fmaxf(a0, a1));       // DPP, no LDS
            const float mn = fmaxf(mi[r], t);
            const float sc = __expf(mi[r] - mn);
            mi[r] = mn;
            const float p0 = __expf(a0 - mn);
            const float p1 = __expf(a1 - mn);
            li[r] = li[r] * sc + (p0 + p1);                 // per-lane partial
#pragma unroll
            for (int nb = 0; nb < 4; ++nb) o[nb][r] *= sc;
            // stash P (bf16) row-major in this wave's LDS patch
            pw[(r + hs * 8) * 32 + l16]      = (bf16_t)p0;
            pw[(r + hs * 8) * 32 + 16 + l16] = (bf16_t)p1;
        }

        // intra-wave LDS RAW: DS ops are in-order per wave; force the wait
        asm volatile("s_wait_dscnt 0x0" ::: "memory");
        const v16bf pa = frag_bf16(pw + l16 * 32 + hs * 8);  // A-layout P

        // ---- O += P @ V  (V stored transposed: (BH, DK, S)) ----
#pragma unroll
        for (int nb = 0; nb < 4; ++nb) {
            const bf16_t* vp =
                VTb + ((size_t)bh * DK_ + nb * 16 + l16) * S_ + kv0 + hs * 8;
            o[nb] = wmma_bf16(pa, frag_bf16(vp), o[nb]);
        }
        asm volatile("" ::: "memory");  // keep next-iter P stores below loads
    }

    // single deferred l reduction per row
    float inv[8];
#pragma unroll
    for (int r = 0; r < 8; ++r)
        inv[r] = 1.0f / fmaxf(rowsum16(li[r]), 1e-20f);

    // merged-head context (row-major (B*S, D)) feeds the output projection
#pragma unroll
    for (int nb = 0; nb < 4; ++nb)
#pragma unroll
        for (int r = 0; r < 8; ++r) {
            const int qrow = q0 + r + hs * 8;
            Ctx[((size_t)(b * S_ + qrow)) * D_ + h * DK_ + nb * 16 + l16] =
                (bf16_t)(o[nb][r] * inv[r]);
        }
}

// ---------------------------------------------------------------------------
extern "C" void kernel_launch(void* const* d_in, const int* in_sizes, int n_in,
                              void* d_out, int out_size, void* d_ws,
                              size_t ws_size, hipStream_t stream) {
    const float* q    = (const float*)d_in[0];
    const float* k    = (const float*)d_in[1];
    const float* v    = (const float*)d_in[2];
    const int*   mask = (const int*)d_in[3];
    const float* Wq   = (const float*)d_in[4];
    const float* bq   = (const float*)d_in[5];
    const float* Wk   = (const float*)d_in[6];
    const float* bk   = (const float*)d_in[7];
    const float* Wv   = (const float*)d_in[8];
    const float* bv   = (const float*)d_in[9];
    const float* Wo   = (const float*)d_in[10];
    const float* bo   = (const float*)d_in[11];

    // workspace carve-up (~55 MB total)
    char*  ws  = (char*)d_ws;
    size_t off = 0;
    auto take = [&](size_t bytes) -> void* {
        off = (off + 255) & ~(size_t)255;
        void* p = ws + off;
        off += bytes;
        return p;
    };
    const size_t wbytes = (size_t)D_ * D_ * sizeof(bf16_t);
    const size_t hbytes = (size_t)BH_ * S_ * DK_ * sizeof(bf16_t);
    bf16_t* Wq_b = (bf16_t*)take(wbytes);
    bf16_t* Wk_b = (bf16_t*)take(wbytes);
    bf16_t* Wv_b = (bf16_t*)take(wbytes);
    bf16_t* Wo_b = (bf16_t*)take(wbytes);
    bf16_t* Qb   = (bf16_t*)take(hbytes);
    bf16_t* Kb   = (bf16_t*)take(hbytes);
    bf16_t* VTb  = (bf16_t*)take(hbytes);
    bf16_t* Ctx  = (bf16_t*)take((size_t)R_ * D_ * sizeof(bf16_t));
    (void)ws_size; (void)in_sizes; (void)n_in; (void)out_size;

    // 1) weights fp32 -> bf16
    const int wn = D_ * D_;
    cvt_f32_to_bf16<<<(wn + 255) / 256, 256, 0, stream>>>(Wq, Wq_b, wn);
    cvt_f32_to_bf16<<<(wn + 255) / 256, 256, 0, stream>>>(Wk, Wk_b, wn);
    cvt_f32_to_bf16<<<(wn + 255) / 256, 256, 0, stream>>>(Wv, Wv_b, wn);
    cvt_f32_to_bf16<<<(wn + 255) / 256, 256, 0, stream>>>(Wo, Wo_b, wn);

    // 2) QKV projections (Q pre-scaled, V transposed)
    dim3 ggrid(R_ / 64, D_ / 64);
    gemm64<0><<<ggrid, 128, 0, stream>>>(q, Wq_b, bq, Qb);
    gemm64<1><<<ggrid, 128, 0, stream>>>(k, Wk_b, bk, Kb);
    gemm64<2><<<ggrid, 128, 0, stream>>>(v, Wv_b, bv, VTb);

    // 3) flash attention
    attn64<<<dim3(S_ / 64, BH_), 128, 0, stream>>>(Qb, Kb, VTb, mask, Ctx);

    // 4) output projection (fp32 out)
    gemm64<3><<<ggrid, 128, 0, stream>>>(Ctx, Wo_b, bo, d_out);
}
